// GroupedQueryAttention_26517128085917
// MI455X (gfx1250) — matrix operand order repaired and checked by reading, not compiled, not run on hardware
//
#include <hip/hip_runtime.h>
#include <stdint.h>

// ---- CDNA5 (gfx1250) wave32 WMMA types ----
typedef __attribute__((ext_vector_type(16))) _Float16 v16h;
typedef __attribute__((ext_vector_type(8)))  float    v8f;
typedef __attribute__((ext_vector_type(4)))  uint32_t u32x4;

#define BB 2
#define SS 2048
#define HID 2048
#define HQ 32
#define HKV 8
#define DD 64
#define MM (BB*SS)   // 4096 total rows

#define EXP2F(x) __builtin_amdgcn_exp2f(x)   // v_exp_f32

union F16Frag { u32x4 u4[2]; uint32_t u[8]; v16h v; };

__device__ inline v16h ld_frag(const _Float16* p0, const _Float16* p1) {
  // p0 -> fragment elements 0..7, p1 -> elements 8..15 (each 16B aligned)
  F16Frag f;
  f.u4[0] = *(const u32x4*)p0;
  f.u4[1] = *(const u32x4*)p1;
  return f.v;
}

__device__ inline v8f wmma_f16(v16h a, v16h b, v8f c) {
  return __builtin_amdgcn_wmma_f32_16x16x32_f16(false, a, false, b, (short)0, c, false, false);
}

// ---------------- fp32 -> fp16 convert ----------------
__global__ void cvt_f16_kernel(const float* __restrict__ in, _Float16* __restrict__ out, int n) {
  int i = blockIdx.x * blockDim.x + threadIdx.x;
  int stride = gridDim.x * blockDim.x;
  for (; i < n; i += stride) out[i] = (_Float16)in[i];
}

// ---- pack weights [K,N] fp32 -> fragment-major f16 pairs:
//      Wf[(kblk*N + col)*16 + t] = pack(W[kblk*32 + 2t][col], W[kblk*32 + 2t + 1][col])
//      so a lane's 8 B-frag regs (t = 4*hf + j%4 + 8*(j/4)) are two 16B chunks.
__global__ void pack_w_kernel(const float* __restrict__ W, uint32_t* __restrict__ Wf,
                              int K, int N) {
  int i = blockIdx.x * blockDim.x + threadIdx.x;
  int n = (K / 2) * N;
  int stride = gridDim.x * blockDim.x;
  for (; i < n; i += stride) {
    int t = i & 15;
    int rest = i >> 4;          // kblk*N + col
    int col = rest % N;
    int kblk = rest / N;
    int k = kblk * 32 + 2 * t;
    union { _Float16 h[2]; uint32_t u; } p;
    p.h[0] = (_Float16)W[(size_t)k * N + col];
    p.h[1] = (_Float16)W[(size_t)(k + 1) * N + col];
    Wf[i] = p.u;
  }
}

// ---------------- WMMA GEMM: C[M,N] = A[M,K](f16) * Wf (fragment-major f16) ----------------
// Wave tile: 32 rows x 64 cols. Block: 8 waves = 256 rows x 64 cols.
// mode 0: store f16 [M,N]
// mode 1: RoPE (per row s = grow % S, head-dim 64) then *scale, store f16 [M,N]
// mode 2: store f32 [M,N]
// mode 3: store f16 transposed per kv-head: Vt[b][hkv][d][s]  (N must be HKV*64)
__global__ __launch_bounds__(256)
void gemm_kernel(const _Float16* __restrict__ A, const uint32_t* __restrict__ Bf,
                 void* __restrict__ C, int N, int K, int mode, float scale)
{
  const int lane = threadIdx.x & 31;
  const int n16  = lane & 15;
  const int hf   = lane >> 4;
  const int wave = threadIdx.x >> 5;
  const int Nbase = blockIdx.x * 64;
  const int Mbase = blockIdx.y * 256 + wave * 32;

  const _Float16* ar0 = A + (size_t)(Mbase + n16) * K + 8 * hf;
  const _Float16* ar1 = ar0 + (size_t)16 * K;
  const u32x4* bp = (const u32x4*)Bf + (size_t)(Nbase + n16) * 4 + hf;
  const size_t bstep = (size_t)N * 4;   // one K-block (32) in u32x4 units

  v8f z = {};
  v8f acc[8];                            // [2*nt + mrow]
#pragma unroll
  for (int i = 0; i < 8; i++) acc[i] = z;

  for (int kb = 0; kb < K; kb += 32) {
    v16h a0 = ld_frag(ar0, ar0 + 16);
    v16h a1 = ld_frag(ar1, ar1 + 16);
    ar0 += 32; ar1 += 32;
#pragma unroll
    for (int nt = 0; nt < 4; nt++) {
      F16Frag bb;
      bb.u4[0] = bp[nt * 64];
      bb.u4[1] = bp[nt * 64 + 2];
      acc[2 * nt]     = wmma_f16(a0, bb.v, acc[2 * nt]);
      acc[2 * nt + 1] = wmma_f16(a1, bb.v, acc[2 * nt + 1]);
    }
    bp += bstep;
  }

#pragma unroll
  for (int mr = 0; mr < 2; mr++) {
#pragma unroll
    for (int r = 0; r < 8; r++) {
      int grow = Mbase + mr * 16 + r + 8 * hf;       // output row (b*S + s)
      float v0 = acc[0 + mr][r], v1 = acc[2 + mr][r];
      float v2 = acc[4 + mr][r], v3 = acc[6 + mr][r];
      if (mode == 1) {
        // RoPE: d = dt*16 + n16 in this 64-wide head tile; inv_freq = exp(-ln(1e4)/32 * (d%32))
        float spos = (float)(grow & (SS - 1));
        float c0, s0, c1, s1;
        sincosf(spos * __expf(-0.28782264f * (float)n16),        &s0, &c0);
        sincosf(spos * __expf(-0.28782264f * (float)(16 + n16)), &s1, &c1);
        float o0 = v0 * c0 - v2 * s0;
        float o1 = v1 * c1 - v3 * s1;
        float o2 = v2 * c0 + v0 * s0;
        float o3 = v3 * c1 + v1 * s1;
        v0 = o0 * scale; v1 = o1 * scale; v2 = o2 * scale; v3 = o3 * scale;
      }
      if (mode == 2) {
        float* crow = (float*)C + (size_t)grow * N + Nbase + n16;
        crow[0] = v0; crow[16] = v1; crow[32] = v2; crow[48] = v3;
      } else if (mode == 3) {
        int b = grow >> 11, s = grow & (SS - 1);
        int hkv = Nbase >> 6;                         // Nbase multiple of 64 -> head fixed
        _Float16* vbase = (_Float16*)C + (size_t)((b * HKV + hkv) * DD) * SS;
        vbase[(size_t)(n16) * SS + s]      = (_Float16)v0;
        vbase[(size_t)(16 + n16) * SS + s] = (_Float16)v1;
        vbase[(size_t)(32 + n16) * SS + s] = (_Float16)v2;
        vbase[(size_t)(48 + n16) * SS + s] = (_Float16)v3;
      } else {
        _Float16* crow = (_Float16*)C + (size_t)grow * N + Nbase + n16;
        crow[0]  = (_Float16)v0; crow[16] = (_Float16)v1;
        crow[32] = (_Float16)v2; crow[48] = (_Float16)v3;
      }
    }
  }
}

// ------------- Flash attention: one wave = 32 queries of one (b,h); K/V frags amortized -------------
// Scores arrive pre-scaled by (1/sqrt(D))*log2(e)  ->  base-2 online softmax (exact).
__global__ __launch_bounds__(128)
void attn_kernel(const _Float16* __restrict__ Qf, const _Float16* __restrict__ Kf,
                 const _Float16* __restrict__ Vt, _Float16* __restrict__ Out)
{
  const int lane = threadIdx.x & 31;
  const int n16  = lane & 15;
  const int hf   = lane >> 4;
  const int gw   = blockIdx.x * 4 + (threadIdx.x >> 5);
  const int qt   = gw & (SS / 32 - 1);     // 64 query tiles of 32
  const int bh   = gw >> 6;
  const int b    = bh >> 5;
  const int h    = bh & (HQ - 1);
  const int hkv  = h >> 2;                 // GQA repeat_interleave by 4
  const int KSTR = HKV * DD;               // 512

  // Q used as B operand of S^T = K_tile x Q^T (elements in table k order over d)
  v16h q0[2], q1[2];
#pragma unroll
  for (int u = 0; u < 2; u++) {
    const _Float16* qrow =
        Qf + (size_t)(b * SS + qt * 32 + u * 16 + n16) * (HQ * DD) + h * DD + 8 * hf;
    q0[u] = ld_frag(qrow,      qrow + 16);   // d 0..31
    q1[u] = ld_frag(qrow + 32, qrow + 48);   // d 32..63
  }

  const _Float16* kr = Kf + (size_t)(b * SS + n16) * KSTR + hkv * DD + 8 * hf;
  const _Float16* vtb = Vt + (size_t)((b * HKV + hkv) * DD) * SS;
  const u32x4* vp0 = (const u32x4*)(vtb + (size_t)(n16) * SS)      + hf;
  const u32x4* vp1 = (const u32x4*)(vtb + (size_t)(16 + n16) * SS) + hf;
  const u32x4* vp2 = (const u32x4*)(vtb + (size_t)(32 + n16) * SS) + hf;
  const u32x4* vp3 = (const u32x4*)(vtb + (size_t)(48 + n16) * SS) + hf;

  v8f z = {};
  v8f o[2][4];
#pragma unroll
  for (int u = 0; u < 2; u++)
#pragma unroll
    for (int dt = 0; dt < 4; dt++) o[u][dt] = z;
  float m[2] = {-3.0e38f, -3.0e38f};
  float l[2] = {0.0f, 0.0f};

  for (int kt = 0; kt < SS / 32; kt++) {
    const _Float16* kr2 = kr + (size_t)16 * KSTR;
    v16h ka0 = ld_frag(kr,       kr + 16);
    v16h ka1 = ld_frag(kr + 32,  kr + 48);
    v16h kb0 = ld_frag(kr2,      kr2 + 16);
    v16h kb1 = ld_frag(kr2 + 32, kr2 + 48);
    kr += (size_t)32 * KSTR;
    __builtin_prefetch(kr, 0, 1);                       // global_prefetch next K tile
    __builtin_prefetch(kr + (size_t)16 * KSTR, 0, 1);

    F16Frag vb0, vb1, vb2, vb3;                         // V^T B-frags, shared by both q-halves
    vb0.u4[0] = vp0[0]; vb0.u4[1] = vp0[2]; vp0 += 4;
    vb1.u4[0] = vp1[0]; vb1.u4[1] = vp1[2]; vp1 += 4;
    vb2.u4[0] = vp2[0]; vb2.u4[1] = vp2[2]; vp2 += 4;
    vb3.u4[0] = vp3[0]; vb3.u4[1] = vp3[2]; vp3 += 4;

#pragma unroll
    for (int u = 0; u < 2; u++) {
      // S^T: rows = keys (r + 8*hf), cols = q (n16)
      v8f s0 = z; s0 = wmma_f16(ka0, q0[u], s0); s0 = wmma_f16(ka1, q1[u], s0);
      v8f s1 = z; s1 = wmma_f16(kb0, q0[u], s1); s1 = wmma_f16(kb1, q1[u], s1);

      float tmax = -3.0e38f;
#pragma unroll
      for (int r = 0; r < 8; r++) tmax = fmaxf(tmax, fmaxf(s0[r], s1[r]));
      tmax = fmaxf(tmax, __shfl_xor(tmax, 16));

      // rescale only when some row's max actually grew (scalar branch: EXEC stays full)
      if (__any(tmax > m[u])) {
        float mnew = fmaxf(m[u], tmax);
        float corr = EXP2F(m[u] - mnew);               // ==1 for rows whose max is unchanged
        m[u] = mnew;
        l[u] *= corr;
#pragma unroll
        for (int r = 0; r < 8; r++) {            // row q = r+8*hf, stat lives in lane r+8*hf
          float cr = __shfl(corr, r + 8 * hf);
          o[u][0][r] *= cr; o[u][1][r] *= cr; o[u][2][r] *= cr; o[u][3][r] *= cr;
        }
      }

      float p0[8], p1[8], rsum = 0.0f;
#pragma unroll
      for (int r = 0; r < 8; r++) {
        p0[r] = EXP2F(s0[r] - m[u]);
        p1[r] = EXP2F(s1[r] - m[u]);
        rsum += p0[r] + p1[r];
      }
      rsum += __shfl_xor(rsum, 16);
      l[u] += rsum;

      v16h pa;                                   // per-lane repack: D-frag row split == A-frag K split
#pragma unroll
      for (int t = 0; t < 8; t++) { pa[t] = (_Float16)p0[t]; pa[8 + t] = (_Float16)p1[t]; }

      o[u][0] = wmma_f16(pa, vb0.v, o[u][0]);
      o[u][1] = wmma_f16(pa, vb1.v, o[u][1]);
      o[u][2] = wmma_f16(pa, vb2.v, o[u][2]);
      o[u][3] = wmma_f16(pa, vb3.v, o[u][3]);
    }
  }

#pragma unroll
  for (int u = 0; u < 2; u++) {
    float linv = 1.0f / l[u];
#pragma unroll
    for (int r = 0; r < 8; r++) {
      float lr = __shfl(linv, r + 8 * hf);
      _Float16* orow =
          Out + (size_t)(b * SS + qt * 32 + u * 16 + r + 8 * hf) * (HQ * DD) + h * DD + n16;
      orow[0]  = (_Float16)(o[u][0][r] * lr);
      orow[16] = (_Float16)(o[u][1][r] * lr);
      orow[32] = (_Float16)(o[u][2][r] * lr);
      orow[48] = (_Float16)(o[u][3][r] * lr);
    }
  }
}

// ---------------- host side ----------------
extern "C" void kernel_launch(void* const* d_in, const int* in_sizes, int n_in,
                              void* d_out, int out_size, void* d_ws, size_t ws_size,
                              hipStream_t stream) {
  (void)in_sizes; (void)n_in; (void)out_size; (void)ws_size;
  const float* X  = (const float*)d_in[0];
  const float* Wq = (const float*)d_in[1];
  const float* Wk = (const float*)d_in[2];
  const float* Wv = (const float*)d_in[3];
  const float* Wo = (const float*)d_in[4];

  char* w = (char*)d_ws;
  _Float16* Xh  = (_Float16*)w; w += (size_t)MM * HID * 2;                 // 16 MB
  uint32_t* WqP = (uint32_t*)w; w += (size_t)(HID / 2) * (HQ * DD) * 4;    //  8 MB
  uint32_t* WkP = (uint32_t*)w; w += (size_t)(HID / 2) * (HKV * DD) * 4;   //  2 MB
  uint32_t* WvP = (uint32_t*)w; w += (size_t)(HID / 2) * (HKV * DD) * 4;   //  2 MB
  uint32_t* WoP = (uint32_t*)w; w += (size_t)((HQ * DD) / 2) * HID * 4;    //  8 MB
  _Float16* Qf  = (_Float16*)w; w += (size_t)MM * HQ * DD * 2;             // 16 MB
  _Float16* Kff = (_Float16*)w; w += (size_t)MM * HKV * DD * 2;            //  4 MB
  _Float16* Vt  = (_Float16*)w; w += (size_t)BB * HKV * DD * SS * 2;       //  4 MB
  _Float16* Af  = (_Float16*)w; w += (size_t)MM * HQ * DD * 2;             // 16 MB

  cvt_f16_kernel<<<2048, 256, 0, stream>>>(X, Xh, MM * HID);
  pack_w_kernel<<<2048, 256, 0, stream>>>(Wq, WqP, HID, HQ * DD);
  pack_w_kernel<<<1024, 256, 0, stream>>>(Wk, WkP, HID, HKV * DD);
  pack_w_kernel<<<1024, 256, 0, stream>>>(Wv, WvP, HID, HKV * DD);
  pack_w_kernel<<<2048, 256, 0, stream>>>(Wo, WoP, HQ * DD, HID);

  // Q = rope(X*Wq) * (1/sqrt(64))*log2(e)  (base-2 softmax); K = rope(X*Wk); V transposed per kv head
  gemm_kernel<<<dim3((HQ * DD) / 64, MM / 256), 256, 0, stream>>>(Xh, WqP, Qf, HQ * DD, HID, 1,
                                                                  0.125f * 1.44269504f);
  gemm_kernel<<<dim3((HKV * DD) / 64, MM / 256), 256, 0, stream>>>(Xh, WkP, Kff, HKV * DD, HID, 1, 1.0f);
  gemm_kernel<<<dim3((HKV * DD) / 64, MM / 256), 256, 0, stream>>>(Xh, WvP, Vt, HKV * DD, HID, 3, 1.0f);

  attn_kernel<<<(BB * HQ * (SS / 32)) / 4, 128, 0, stream>>>(Qf, Kff, Vt, Af);

  gemm_kernel<<<dim3(HID / 64, MM / 256), 256, 0, stream>>>(Af, WoP, d_out, HID, HQ * DD, 2, 1.0f);
}
